// GCN2_attention_89008902243178
// MI455X (gfx1250) — matrix operand order repaired and checked by reading, not compiled, not run on hardware
//
#include <hip/hip_runtime.h>
#include <hip/hip_bf16.h>

// ---------------------------------------------------------------------------
// 2-layer GCN (PyG GCNConv semantics) for MI455X / gfx1250.
//  * GEMMs via v_wmma_f32_16x16x32_bf16 (f32 accumulate), W^T staged in LDS.
//  * Edge aggregation via wave-per-edge global_atomic_add_f32 (targets fit L2).
// ---------------------------------------------------------------------------

typedef __attribute__((ext_vector_type(16))) __bf16 v16bf;
typedef __attribute__((ext_vector_type(8)))  float  v8f;

#define N_NODES   100000
#define N_EDGES   1600000
#define F_IN      128
#define F_HID     128
#define N_CLASSES 40
#define N_CLS_PAD 48
#define BPAD      136   // padded LDS row length (bf16 elems) -> 272B row stride

// ---------------- graph-normalization kernels ------------------------------

__global__ void k_deg_init(float* deg) {
  int i = blockIdx.x * 256 + threadIdx.x;
  if (i < N_NODES) deg[i] = 1.0f;               // self-loop weight 1
}

__global__ void k_deg_acc(const int* __restrict__ ei, const float* __restrict__ w,
                          float* deg) {
  int e = blockIdx.x * 256 + threadIdx.x;
  if (e < N_EDGES) atomicAdd(&deg[ei[N_EDGES + e]], w[e]);   // col = ei[1][e]
}

__global__ void k_dinv(float* deg) {
  int i = blockIdx.x * 256 + threadIdx.x;
  if (i < N_NODES) {
    float d = deg[i];
    deg[i] = (d > 0.0f) ? rsqrtf(d) : 0.0f;
  }
}

__global__ void k_norm(const int* __restrict__ ei, const float* __restrict__ w,
                       const float* __restrict__ dinv, float* __restrict__ nrm) {
  int e = blockIdx.x * 256 + threadIdx.x;
  if (e < N_EDGES) nrm[e] = dinv[ei[e]] * w[e] * dinv[ei[N_EDGES + e]];
}

// ---------------- weight transpose + f32->bf16 conversion ------------------

__global__ void k_w1t(const float* __restrict__ W1, __bf16* __restrict__ w1t) {
  int t = blockIdx.x * 256 + threadIdx.x;       // t = n*128 + k
  if (t >= F_HID * F_IN) return;
  int n = t >> 7, k = t & 127;
  w1t[t] = (__bf16)W1[(size_t)k * F_HID + n];
}

__global__ void k_w2t(const float* __restrict__ W2, __bf16* __restrict__ w2t) {
  int t = blockIdx.x * 256 + threadIdx.x;       // t = n*128 + k, n in [0,48)
  if (t >= N_CLS_PAD * F_HID) return;
  int n = t >> 7, k = t & 127;
  w2t[t] = (n < N_CLASSES) ? (__bf16)W2[(size_t)k * N_CLASSES + n] : (__bf16)0.0f;
}

// ---------------- GEMM1: XW = bf16(X) @ bf16(W1), f32 accumulate -----------

__global__ __launch_bounds__(256)
void k_gemm1(const float* __restrict__ x, const __bf16* __restrict__ w1t,
             float* __restrict__ xw) {
  __shared__ __bf16 bs[F_HID][BPAD];
  int tid = threadIdx.x;
  // stage W1^T (bf16, [n][k]) into LDS, 8B chunks
  for (int c = tid; c < F_HID * (F_IN / 4); c += 256) {
    int n = c >> 5, k4 = (c & 31) * 4;
    *(uint2*)&bs[n][k4] = *(const uint2*)(w1t + (size_t)n * F_IN + k4);
  }
  __syncthreads();

  int wave = tid >> 5, lane = tid & 31;
  int strip = blockIdx.x * 8 + wave;
  if (strip >= N_NODES / 16) return;            // wave-uniform: EXEC stays all-1
  int m0 = strip * 16;
  int h  = lane >> 4;
  int mrow = m0 + (lane & 15);
  const float* xrow = x + (size_t)mrow * F_IN;

  // A fragments for 4 k-steps (convert f32 -> bf16 in-register)
  v16bf a[4];
#pragma unroll
  for (int j = 0; j < 4; ++j) {
    int kb = 32 * j + 16 * h;
#pragma unroll
    for (int t = 0; t < 8; ++t) {
      a[j][t]     = (__bf16)xrow[kb + t];
      a[j][8 + t] = (__bf16)xrow[kb + 8 + t];
    }
  }

  int ncol = lane & 15;
#pragma unroll
  for (int nt = 0; nt < 8; ++nt) {
    v8f acc = {0.f, 0.f, 0.f, 0.f, 0.f, 0.f, 0.f, 0.f};
#pragma unroll
    for (int j = 0; j < 4; ++j) {
      const __bf16* bp = &bs[nt * 16 + ncol][32 * j + 16 * h];
      v16bf b;
#pragma unroll
      for (int t = 0; t < 16; ++t) b[t] = bp[t];
      acc = __builtin_amdgcn_wmma_f32_16x16x32_bf16(false, a[j], false, b,
                                                    (short)0, acc, false, false);
    }
    float* orow = xw + (size_t)(m0 + 8 * h) * F_HID + nt * 16 + ncol;
#pragma unroll
    for (int r = 0; r < 8; ++r) orow[(size_t)r * F_HID] = acc[r];
  }
}

// ---------------- layer-1 aggregation --------------------------------------

__global__ void k_agg1_init(const float* __restrict__ dinv,
                            const float* __restrict__ xw, float* __restrict__ H) {
  int t = blockIdx.x * 256 + threadIdx.x;       // N_NODES*32 threads
  if (t >= N_NODES * 32) return;
  int i = t >> 5, c4 = (t & 31) * 4;
  float s = dinv[i]; s = s * s;                 // self-loop norm
  float4 v = *(const float4*)(xw + (size_t)i * F_HID + c4);
  float4 o = {s * v.x, s * v.y, s * v.z, s * v.w};
  *(float4*)(H + (size_t)i * F_HID + c4) = o;
}

__global__ void k_agg1_edges(const int* __restrict__ ei, const float* __restrict__ nrm,
                             const float* __restrict__ xw, float* __restrict__ H) {
  int t = blockIdx.x * 256 + threadIdx.x;       // N_EDGES*32 threads, wave/edge
  if (t >= N_EDGES * 32) return;
  int e = t >> 5, c4 = (t & 31) * 4;
  int r = ei[e], c = ei[N_EDGES + e];
  float nm = nrm[e];
  float4 v = *(const float4*)(xw + (size_t)r * F_HID + c4);
  float* dst = H + (size_t)c * F_HID + c4;
  atomicAdd(dst + 0, nm * v.x);
  atomicAdd(dst + 1, nm * v.y);
  atomicAdd(dst + 2, nm * v.z);
  atomicAdd(dst + 3, nm * v.w);
}

__global__ void k_relu_bf16(const float* __restrict__ H, const float* __restrict__ b1,
                            __bf16* __restrict__ hb) {
  int t = blockIdx.x * 256 + threadIdx.x;
  if (t >= N_NODES * 32) return;
  int i = t >> 5, c4 = (t & 31) * 4;
  float4 v = *(const float4*)(H + (size_t)i * F_HID + c4);
  float4 b = *(const float4*)(b1 + c4);
  __bf16* dst = hb + (size_t)i * F_HID + c4;
  dst[0] = (__bf16)fmaxf(v.x + b.x, 0.0f);
  dst[1] = (__bf16)fmaxf(v.y + b.y, 0.0f);
  dst[2] = (__bf16)fmaxf(v.z + b.z, 0.0f);
  dst[3] = (__bf16)fmaxf(v.w + b.w, 0.0f);
}

// ---------------- GEMM2: HW2 = h_bf16 @ bf16(W2) ---------------------------

__global__ __launch_bounds__(256)
void k_gemm2(const __bf16* __restrict__ hb, const __bf16* __restrict__ w2t,
             float* __restrict__ hw2) {
  __shared__ __bf16 bs[N_CLS_PAD][BPAD];
  int tid = threadIdx.x;
  for (int c = tid; c < N_CLS_PAD * (F_HID / 4); c += 256) {
    int n = c >> 5, k4 = (c & 31) * 4;
    *(uint2*)&bs[n][k4] = *(const uint2*)(w2t + (size_t)n * F_HID + k4);
  }
  __syncthreads();

  int wave = tid >> 5, lane = tid & 31;
  int strip = blockIdx.x * 8 + wave;
  if (strip >= N_NODES / 16) return;
  int m0 = strip * 16;
  int h  = lane >> 4;
  const __bf16* hrow = hb + (size_t)(m0 + (lane & 15)) * F_HID;

  v16bf a[4];
#pragma unroll
  for (int j = 0; j < 4; ++j) {
    int kb = 32 * j + 16 * h;
#pragma unroll
    for (int t = 0; t < 8; ++t) {
      a[j][t]     = hrow[kb + t];
      a[j][8 + t] = hrow[kb + 8 + t];
    }
  }

  int ncol = lane & 15;
#pragma unroll
  for (int nt = 0; nt < 3; ++nt) {
    v8f acc = {0.f, 0.f, 0.f, 0.f, 0.f, 0.f, 0.f, 0.f};
#pragma unroll
    for (int j = 0; j < 4; ++j) {
      const __bf16* bp = &bs[nt * 16 + ncol][32 * j + 16 * h];
      v16bf b;
#pragma unroll
      for (int t = 0; t < 16; ++t) b[t] = bp[t];
      acc = __builtin_amdgcn_wmma_f32_16x16x32_bf16(false, a[j], false, b,
                                                    (short)0, acc, false, false);
    }
    int col = nt * 16 + ncol;
    if (col < N_CLASSES) {
      float* orow = hw2 + (size_t)(m0 + 8 * h) * N_CLASSES + col;
#pragma unroll
      for (int r = 0; r < 8; ++r) orow[(size_t)r * N_CLASSES] = acc[r];
    }
  }
}

// ---------------- layer-2 aggregation + softmax -----------------------------

__global__ void k_agg2_init(const float* __restrict__ dinv,
                            const float* __restrict__ hw2, float* __restrict__ O2) {
  int t = blockIdx.x * 256 + threadIdx.x;       // N_NODES*40
  if (t >= N_NODES * N_CLASSES) return;
  int i = t / N_CLASSES;
  float s = dinv[i];
  O2[t] = s * s * hw2[t];
}

__global__ void k_agg2_edges(const int* __restrict__ ei, const float* __restrict__ nrm,
                             const float* __restrict__ hw2, float* __restrict__ O2) {
  int t = blockIdx.x * 256 + threadIdx.x;       // N_EDGES*40
  if (t >= N_EDGES * N_CLASSES) return;
  int e = t / N_CLASSES;
  int c = t - e * N_CLASSES;
  int r = ei[e], cc = ei[N_EDGES + e];
  atomicAdd(&O2[(size_t)cc * N_CLASSES + c], nrm[e] * hw2[(size_t)r * N_CLASSES + c]);
}

__global__ __launch_bounds__(256)
void k_softmax(const float* __restrict__ O2, const float* __restrict__ b2,
               float* __restrict__ out) {
  int wave = threadIdx.x >> 5, lane = threadIdx.x & 31;
  int row = blockIdx.x * 8 + wave;
  if (row >= N_NODES) return;
  const float* p = O2 + (size_t)row * N_CLASSES;
  float v0 = p[lane] + b2[lane];
  float v1 = (lane < 8) ? (p[32 + lane] + b2[32 + lane]) : -3.402823466e38f;
  float m = fmaxf(v0, v1);
#pragma unroll
  for (int off = 16; off > 0; off >>= 1) m = fmaxf(m, __shfl_xor(m, off, 32));
  float e0 = __expf(v0 - m);
  float e1 = (lane < 8) ? __expf(v1 - m) : 0.0f;
  float s = e0 + e1;
#pragma unroll
  for (int off = 16; off > 0; off >>= 1) s += __shfl_xor(s, off, 32);
  float inv = 1.0f / s;
  out[(size_t)row * N_CLASSES + lane] = e0 * inv;
  if (lane < 8) out[(size_t)row * N_CLASSES + 32 + lane] = e1 * inv;
}

// ---------------- launcher ---------------------------------------------------

extern "C" void kernel_launch(void* const* d_in, const int* in_sizes, int n_in,
                              void* d_out, int out_size, void* d_ws, size_t ws_size,
                              hipStream_t stream) {
  const float* x  = (const float*)d_in[0];
  const int*   ei = (const int*)d_in[1];
  const float* ew = (const float*)d_in[2];
  // d_in[3] = attention: softmax over a single element == 1.0, drops out.
  const float* W1 = (const float*)d_in[4];
  const float* b1 = (const float*)d_in[5];
  const float* W2 = (const float*)d_in[6];
  const float* b2 = (const float*)d_in[7];
  float* out = (float*)d_out;

  // workspace layout (128B-aligned offsets)
  char* ws = (char*)d_ws;
  const size_t OFF_DINV = 0;                       // 400,000 B (deg -> dinv in place)
  const size_t OFF_NORM = 400384;                  // 6,400,000 B
  const size_t OFF_XW   = OFF_NORM + 6400000;      // 51,200,000 B (reused as HW2)
  const size_t OFF_H    = OFF_XW + 51200000;       // 51,200,000 B (reused as O2)
  const size_t OFF_HB   = OFF_H + 51200000;        // 25,600,000 B (bf16 h)
  const size_t OFF_W1T  = OFF_HB + 25600000;       // 32,768 B
  const size_t OFF_W2T  = OFF_W1T + 32768;         // 12,288 B

  float*  dinv = (float*)(ws + OFF_DINV);
  float*  nrm  = (float*)(ws + OFF_NORM);
  float*  XW   = (float*)(ws + OFF_XW);
  float*  H    = (float*)(ws + OFF_H);
  __bf16* hb   = (__bf16*)(ws + OFF_HB);
  __bf16* w1t  = (__bf16*)(ws + OFF_W1T);
  __bf16* w2t  = (__bf16*)(ws + OFF_W2T);
  float*  HW2  = XW;    // reuse (layer-1 XW dead after relu)
  float*  O2   = H;     // reuse (layer-1 H dead after relu)

  // weights: transpose + convert
  k_w1t<<<(F_HID * F_IN + 255) / 256, 256, 0, stream>>>(W1, w1t);
  k_w2t<<<(N_CLS_PAD * F_HID + 255) / 256, 256, 0, stream>>>(W2, w2t);

  // normalization
  k_deg_init<<<(N_NODES + 255) / 256, 256, 0, stream>>>(dinv);
  k_deg_acc<<<N_EDGES / 256, 256, 0, stream>>>(ei, ew, dinv);
  k_dinv<<<(N_NODES + 255) / 256, 256, 0, stream>>>(dinv);
  k_norm<<<N_EDGES / 256, 256, 0, stream>>>(ei, ew, dinv, nrm);

  // layer 1
  k_gemm1<<<(N_NODES / 16 + 7) / 8, 256, 0, stream>>>(x, w1t, XW);
  k_agg1_init<<<(N_NODES * 32) / 256, 256, 0, stream>>>(dinv, XW, H);
  k_agg1_edges<<<(N_EDGES * 32) / 256, 256, 0, stream>>>(ei, nrm, XW, H);
  k_relu_bf16<<<(N_NODES * 32) / 256, 256, 0, stream>>>(H, b1, hb);

  // layer 2
  k_gemm2<<<(N_NODES / 16 + 7) / 8, 256, 0, stream>>>(hb, w2t, HW2);
  k_agg2_init<<<(N_NODES * N_CLASSES) / 256, 256, 0, stream>>>(dinv, HW2, O2);
  k_agg2_edges<<<(N_EDGES * N_CLASSES) / 256, 256, 0, stream>>>(ei, nrm, HW2, O2);

  // softmax -> d_out
  k_softmax<<<(N_NODES + 7) / 8, 256, 0, stream>>>(O2, b2, out);
}
